// _RPN_29205777613278
// MI455X (gfx1250) — matrix operand order repaired and checked by reading, not compiled, not run on hardware
//
#include <hip/hip_runtime.h>
#include <hip/hip_bf16.h>

typedef __attribute__((ext_vector_type(4)))  unsigned int u32x4;
typedef __attribute__((ext_vector_type(16))) __bf16       v16bf;
typedef __attribute__((ext_vector_type(8)))  float        v8f;

// ---------------- helpers ----------------
__device__ __forceinline__ unsigned short f2bf(float f) {
    unsigned int u = __builtin_bit_cast(unsigned int, f);
    unsigned int r = u + 0x7FFFu + ((u >> 16) & 1u);   // round-to-nearest-even
    return (unsigned short)(r >> 16);
}
__device__ __forceinline__ v16bf make_frag(u32x4 lo, u32x4 hi) {
    union { u32x4 q[2]; v16bf v; } u;
    u.q[0] = lo; u.q[1] = hi;
    return u.v;
}

// Problem constants
#define MTOT   38400      // B*T*H*W
#define SPB    4800       // T*H*W
#define CIN    512
#define KTOT   13824      // 512*27
#define NCLS   36
#define NBBX   108
#define NHEAD  144
#define NA     18
#define NANCH  86400      // SPB*NA
#define TOPN   300

// ---------------- 1) weight prepack ----------------
// Wpack[n][koff*512 + c] = W_conv[n][c][koff]  (bf16, contiguous K per output channel)
__global__ void __launch_bounds__(256) prepack_conv_w(const float* __restrict__ w,
                                                      unsigned short* __restrict__ wpack) {
    int idx = blockIdx.x * 256 + threadIdx.x;          // < 512*13824
    int n = idx / KTOT;
    int rem = idx - n * KTOT;
    int koff = rem >> 9;
    int c = rem & 511;
    wpack[idx] = f2bf(w[(size_t)(n * 512 + c) * 27 + koff]);
}

// W2pack[n][c], n: 0..35 = W_cls rows, 36..143 = W_bbox rows (bf16)
__global__ void __launch_bounds__(256) prepack_head_w(const float* __restrict__ wc,
                                                      const float* __restrict__ wb,
                                                      unsigned short* __restrict__ w2) {
    int idx = blockIdx.x * 256 + threadIdx.x;          // < 144*512
    int n = idx >> 9;
    int c = idx & 511;
    float v = (n < NCLS) ? wc[n * 512 + c] : wb[(n - NCLS) * 512 + c];
    w2[idx] = f2bf(v);
}

// ---------------- 2) 3x3x3 conv + bias + ReLU as WMMA implicit GEMM ----------------
// grid (300, 8): block computes 128(M) x 64(N) tile; 8 waves -> 4(M) x 2(N), each 32x32.
__global__ void __launch_bounds__(256) conv3d_wmma(const float* __restrict__ in,
                                                   const unsigned short* __restrict__ wpack,
                                                   const float* __restrict__ bias,
                                                   unsigned short* __restrict__ rpn) {
    __shared__ __align__(16) unsigned short Abuf[128][40];   // 32 cols + pad

    const int tid = threadIdx.x;
    const int m0 = blockIdx.x * 128;
    const int n0 = blockIdx.y * 64;

    // staging assignment: fixed row per thread, columns cBase + 2*i
    const int mLocal = tid & 127;
    const int cBase  = tid >> 7;                       // 0 or 1
    const int mg = m0 + mLocal;
    const int bIdx = mg / SPB;
    int r  = mg - bIdx * SPB;
    const int t = r / 300;   r  -= t * 300;
    const int h = r / 20;    const int w = r - h * 20;

    const int lane   = tid & 31;
    const int waveId = tid >> 5;
    const int half   = lane >> 4;
    const int lrow   = lane & 15;
    const int mt = (waveId & 3) * 32;
    const int nt = (waveId >> 2) * 32;

    v8f acc00 = {0.f,0.f,0.f,0.f,0.f,0.f,0.f,0.f};
    v8f acc01 = acc00, acc10 = acc00, acc11 = acc00;

    const unsigned short* brow0 = wpack + (size_t)(n0 + nt + lrow) * KTOT;
    const unsigned short* brow1 = wpack + (size_t)(n0 + nt + 16 + lrow) * KTOT;

    for (int kc = 0; kc < 432; ++kc) {
        const int koff = kc >> 4;
        const int c0   = (kc & 15) << 5;
        const int dt = koff / 9;
        const int rm = koff - dt * 9;
        const int dh = rm / 3;
        const int dw = rm - dh * 3;
        const int tp = t + dt - 1, hp = h + dh - 1, wp = w + dw - 1;
        const bool valid = ((unsigned)tp < 16u) && ((unsigned)hp < 15u) && ((unsigned)wp < 20u);
        const float* src = in + ((((long long)bIdx * 512 + (c0 + cBase)) * 16 + tp) * 300
                                 + (long long)hp * 20 + wp);

        __syncthreads();                                // previous compute done
        #pragma unroll
        for (int i = 0; i < 16; ++i) {
            float v = valid ? src[(long long)9600 * i] : 0.0f;   // 2 channels stride
            Abuf[mLocal][cBase + 2 * i] = f2bf(v);
        }
        __syncthreads();

        const int kb = kc * 32;
        // A fragments (documented 16-bit A 16x32 layout)
        const int ka = half * 8;
        v16bf A0 = make_frag(*(const u32x4*)&Abuf[mt + lrow][ka],
                             *(const u32x4*)&Abuf[mt + lrow][ka + 16]);
        v16bf A1 = make_frag(*(const u32x4*)&Abuf[mt + 16 + lrow][ka],
                             *(const u32x4*)&Abuf[mt + 16 + lrow][ka + 16]);
        // B fragments: same K order as A (lanes 0-15: K 0..7,16..23; lanes 16-31: K 8..15,24..31)
        const unsigned short* bp0 = brow0 + kb + half * 8;
        const unsigned short* bp1 = brow1 + kb + half * 8;
        __builtin_prefetch(brow0 + kb + 32, 0, 1);
        v16bf B0 = make_frag(*(const u32x4*)bp0, *(const u32x4*)(bp0 + 16));
        v16bf B1 = make_frag(*(const u32x4*)bp1, *(const u32x4*)(bp1 + 16));

        acc00 = __builtin_amdgcn_wmma_f32_16x16x32_bf16(false, A0, false, B0, (short)0, acc00, false, false);
        acc01 = __builtin_amdgcn_wmma_f32_16x16x32_bf16(false, A0, false, B1, (short)0, acc01, false, false);
        acc10 = __builtin_amdgcn_wmma_f32_16x16x32_bf16(false, A1, false, B0, (short)0, acc10, false, false);
        acc11 = __builtin_amdgcn_wmma_f32_16x16x32_bf16(false, A1, false, B1, (short)0, acc11, false, false);
    }

    // bias + ReLU, store bf16 [m][512]
    const float bv0 = bias[n0 + nt + lrow];
    const float bv1 = bias[n0 + nt + 16 + lrow];
    auto storeTile = [&](v8f a, int mtile, int ntile, float bv) {
        const int gmB = m0 + mtile + 8 * half;
        const int gn  = n0 + ntile + lrow;
        #pragma unroll
        for (int rr = 0; rr < 8; ++rr) {
            float v = a[rr] + bv;
            v = v > 0.f ? v : 0.f;
            rpn[(size_t)(gmB + rr) * 512 + gn] = f2bf(v);
        }
    };
    storeTile(acc00, mt,      nt,      bv0);
    storeTile(acc01, mt,      nt + 16, bv1);
    storeTile(acc10, mt + 16, nt,      bv0);
    storeTile(acc11, mt + 16, nt + 16, bv1);
}

// ---------------- 3) fused 1x1 cls+bbox GEMM (M=38400, N=144, K=512) ----------------
// one 16x16 WMMA tile per wave; grid 2700 * 8 waves = 21600 tiles = 2400(M) x 9(N)
__global__ void __launch_bounds__(256) gemm1x1_wmma(const unsigned short* __restrict__ rpn,
                                                    const unsigned short* __restrict__ w2,
                                                    float* __restrict__ out2) {
    const int wt   = blockIdx.x * 8 + (threadIdx.x >> 5);
    const int lane = threadIdx.x & 31;
    const int half = lane >> 4;
    const int lrow = lane & 15;
    const int mT = wt / 9;
    const int nT = wt - mT * 9;
    const int m0 = mT * 16;
    const int n0 = nT * 16;

    const unsigned short* arow = rpn + (size_t)(m0 + lrow) * 512;
    const unsigned short* brow = w2  + (size_t)(n0 + lrow) * 512;

    v8f acc = {0.f,0.f,0.f,0.f,0.f,0.f,0.f,0.f};
    for (int k0 = 0; k0 < 512; k0 += 32) {
        const unsigned short* ap = arow + k0 + half * 8;
        const unsigned short* bp = brow + k0 + half * 8;
        v16bf A  = make_frag(*(const u32x4*)ap, *(const u32x4*)(ap + 16));
        v16bf Bf = make_frag(*(const u32x4*)bp, *(const u32x4*)(bp + 16));
        acc = __builtin_amdgcn_wmma_f32_16x16x32_bf16(false, A, false, Bf, (short)0, acc, false, false);
    }
    #pragma unroll
    for (int rr = 0; rr < 8; ++rr)
        out2[(size_t)(m0 + rr + 8 * half) * NHEAD + n0 + lrow] = acc[rr];
}

// ---------------- 4) softmax / bbox-transform epilogue ----------------
__global__ void __launch_bounds__(256) epilogue_kernel(const float* __restrict__ g2,
                                                       const float* __restrict__ b_cls,
                                                       const float* __restrict__ b_bbox,
                                                       const float* __restrict__ anchors,
                                                       const float* __restrict__ im_info,
                                                       float* __restrict__ out,
                                                       float* __restrict__ scores,
                                                       float* __restrict__ props) {
    const int m = blockIdx.x * 256 + threadIdx.x;      // < 38400
    const int b = m / SPB;
    const int s = m - b * SPB;
    const float* row = g2 + (size_t)m * NHEAD;
    float* probOut = out + (size_t)TOPN * 8 * 7;                       // 16800
    float* bboxOut = probOut + (size_t)8 * NCLS * SPB;                 // + 1382400

    const float limx = im_info[b * 3 + 1] - 1.0f;
    const float limy = im_info[b * 3 + 0] - 1.0f;
    const float limt = 15.0f;

    for (int a = 0; a < NA; ++a) {
        // paired softmax: channel a vs channel a+18 at same (t,h,w)
        float x0 = row[a] + b_cls[a];
        float x1 = row[NA + a] + b_cls[NA + a];
        float mx = fmaxf(x0, x1);
        float e0 = __expf(x0 - mx), e1 = __expf(x1 - mx);
        float inv = 1.0f / (e0 + e1);
        float p0 = e0 * inv, p1 = e1 * inv;
        probOut[((size_t)b * NCLS + a) * SPB + s]      = p0;
        probOut[((size_t)b * NCLS + NA + a) * SPB + s] = p1;
        scores[(size_t)b * NANCH + s * NA + a] = p1;

        // bbox deltas (raw bbox_pred output in NCDHW)
        const float* d = row + NCLS + a * 6;
        float dx = d[0] + b_bbox[a * 6 + 0];
        float dy = d[1] + b_bbox[a * 6 + 1];
        float dz = d[2] + b_bbox[a * 6 + 2];
        float dw = d[3] + b_bbox[a * 6 + 3];
        float dh = d[4] + b_bbox[a * 6 + 4];
        float dl = d[5] + b_bbox[a * 6 + 5];
        bboxOut[((size_t)b * NBBX + a * 6 + 0) * SPB + s] = dx;
        bboxOut[((size_t)b * NBBX + a * 6 + 1) * SPB + s] = dy;
        bboxOut[((size_t)b * NBBX + a * 6 + 2) * SPB + s] = dz;
        bboxOut[((size_t)b * NBBX + a * 6 + 3) * SPB + s] = dw;
        bboxOut[((size_t)b * NBBX + a * 6 + 4) * SPB + s] = dh;
        bboxOut[((size_t)b * NBBX + a * 6 + 5) * SPB + s] = dl;

        // bbox_transform_inv + clip
        const float* an = anchors + (size_t)(s * NA + a) * 6;
        float aw = an[3] - an[0] + 1.0f;
        float ah = an[4] - an[1] + 1.0f;
        float al = an[5] - an[2] + 1.0f;
        float cx = an[0] + 0.5f * aw;
        float cy = an[1] + 0.5f * ah;
        float ct = an[2] + 0.5f * al;
        float pcx = dx * aw + cx, pcy = dy * ah + cy, pct = dz * al + ct;
        float pw = __expf(dw) * aw, ph = __expf(dh) * ah, pl = __expf(dl) * al;
        float q0 = fminf(fmaxf(pcx - 0.5f * pw, 0.f), limx);
        float q1 = fminf(fmaxf(pcy - 0.5f * ph, 0.f), limy);
        float q2 = fminf(fmaxf(pct - 0.5f * pl, 0.f), limt);
        float q3 = fminf(fmaxf(pcx + 0.5f * pw, 0.f), limx);
        float q4 = fminf(fmaxf(pcy + 0.5f * ph, 0.f), limy);
        float q5 = fminf(fmaxf(pct + 0.5f * pl, 0.f), limt);
        float* pp = props + ((size_t)b * NANCH + s * NA + a) * 6;
        pp[0] = q0; pp[1] = q1; pp[2] = q2; pp[3] = q3; pp[4] = q4; pp[5] = q5;
    }
}

// ---------------- 5) top-300 per batch: iterative argmax + mask ----------------
__global__ void __launch_bounds__(1024) topk_kernel(float* __restrict__ scores,
                                                    const float* __restrict__ props,
                                                    float* __restrict__ rois) {
    __shared__ float sv[1024];
    __shared__ int   si[1024];
    const int b = blockIdx.x;
    const int tid = threadIdx.x;
    float* sc = scores + (size_t)b * NANCH;

    for (int r = 0; r < TOPN; ++r) {
        float best = -3.4e38f;
        int bi = 0x7fffffff;
        for (int i = tid; i < NANCH; i += 1024) {
            float v = sc[i];
            if (v > best || (v == best && i < bi)) { best = v; bi = i; }
        }
        sv[tid] = best; si[tid] = bi;
        __syncthreads();
        for (int sft = 512; sft > 0; sft >>= 1) {
            if (tid < sft) {
                float ov = sv[tid + sft]; int oi = si[tid + sft];
                if (ov > sv[tid] || (ov == sv[tid] && oi < si[tid])) { sv[tid] = ov; si[tid] = oi; }
            }
            __syncthreads();
        }
        if (tid == 0) {
            int idx = si[0];
            const float* p = props + ((size_t)b * NANCH + idx) * 6;
            float* o = rois + ((size_t)b * TOPN + r) * 7;
            o[0] = (float)b;
            #pragma unroll
            for (int j = 0; j < 6; ++j) o[1 + j] = p[j];
            sc[idx] = -3.4e38f;
        }
        __syncthreads();
    }
}

// ---------------- launch ----------------
extern "C" void kernel_launch(void* const* d_in, const int* in_sizes, int n_in,
                              void* d_out, int out_size, void* d_ws, size_t ws_size,
                              hipStream_t stream) {
    const float* base_feat = (const float*)d_in[0];
    const float* im_info   = (const float*)d_in[1];
    const float* W_conv    = (const float*)d_in[2];
    const float* b_conv    = (const float*)d_in[3];
    const float* W_cls     = (const float*)d_in[4];
    const float* b_cls     = (const float*)d_in[5];
    const float* W_bbox    = (const float*)d_in[6];
    const float* b_bbox    = (const float*)d_in[7];
    const float* anchors   = (const float*)d_in[8];
    float* out = (float*)d_out;

    char* ws = (char*)d_ws;
    unsigned short* wpack = (unsigned short*)(ws);                  // 14,155,776 B
    unsigned short* w2    = (unsigned short*)(ws + 14155776);       //    147,456 B
    unsigned short* rpn   = (unsigned short*)(ws + 14303232);       // 39,321,600 B
    float*          g2    = (float*)(ws + 53624832);                // 22,118,400 B
    float*          sc    = (float*)(ws + 75743232);                //  2,764,800 B
    float*          pr    = (float*)(ws + 78508032);                // 16,588,800 B

    prepack_conv_w<<<27648, 256, 0, stream>>>(W_conv, wpack);
    prepack_head_w<<<288, 256, 0, stream>>>(W_cls, W_bbox, w2);
    conv3d_wmma<<<dim3(300, 8), 256, 0, stream>>>(base_feat, wpack, b_conv, rpn);
    gemm1x1_wmma<<<2700, 256, 0, stream>>>(rpn, w2, g2);
    epilogue_kernel<<<150, 256, 0, stream>>>(g2, b_cls, b_bbox, anchors, im_info, out, sc, pr);
    topk_kernel<<<8, 1024, 0, stream>>>(sc, pr, out);
    (void)in_sizes; (void)n_in; (void)out_size; (void)ws_size;
}
